// ExampleGNN_5643587027283
// MI455X (gfx1250) — matrix operand order repaired and checked by reading, not compiled, not run on hardware
//
#include <hip/hip_runtime.h>

#define ND 128          // feature dim, fixed by the reference
#define NEG_SLOPE 0.2f

typedef __attribute__((ext_vector_type(16))) __bf16 v16bf;
typedef __attribute__((ext_vector_type(8)))  float  v8f;

__device__ __forceinline__ unsigned short f2bf(float f) {
    unsigned u = __float_as_uint(f);
    return (unsigned short)((u + 0x7FFFu + ((u >> 16) & 1u)) >> 16);  // RNE
}
__device__ __forceinline__ void put2(v16bf& v, int i, unsigned w) {
    v[i]     = __builtin_bit_cast(__bf16, (unsigned short)(w & 0xFFFFu));
    v[i + 1] = __builtin_bit_cast(__bf16, (unsigned short)(w >> 16));
}

// ---------------------------------------------------------------------------
// GEMM: OUT[nrows x ncols] = X[nrows x 128] * W[128 x ncols] (+bias)
// bf16 operands, f32 accumulate via v_wmma_f32_16x16x32_bf16.
// Block = 128 threads = 4 waves; block tile = 64 rows x (NT*16) cols.
// ---------------------------------------------------------------------------
template <int NT>
__global__ __launch_bounds__(128) void gemm_bf16_wmma(
    const float* __restrict__ X, const float* __restrict__ W,
    const float* __restrict__ bias, float* __restrict__ OUT,
    int nrows, int ncols)
{
    __shared__ unsigned short Wt[NT * 16][ND];  // W transposed, bf16
    __shared__ unsigned short Al[64][ND];       // X tile, bf16

    const int tid = threadIdx.x;
    const int row0 = blockIdx.x * 64;

    // Stage W^T into LDS (zero-pad columns >= ncols)
    for (int idx = tid; idx < NT * 16 * ND; idx += 128) {
        int k = idx / (NT * 16);
        int n = idx - k * (NT * 16);
        float v = (n < ncols) ? W[k * ncols + n] : 0.0f;
        Wt[n][k] = f2bf(v);
    }
    // Stage X tile into LDS (zero-pad rows >= nrows)
    for (int idx = tid; idx < 64 * ND; idx += 128) {
        int r = idx >> 7, k = idx & 127;
        int row = row0 + r;
        Al[r][k] = (row < nrows) ? f2bf(X[(size_t)row * ND + k]) : (unsigned short)0;
    }
    __syncthreads();

    const int wave = tid >> 5, lane = tid & 31;
    const int m = lane & 15, hl = lane >> 4;

    v8f acc[NT];
    v8f zero = {0.f, 0.f, 0.f, 0.f, 0.f, 0.f, 0.f, 0.f};
#pragma unroll
    for (int t = 0; t < NT; ++t) acc[t] = zero;

#pragma unroll
    for (int ks = 0; ks < 4; ++ks) {
        const int K0 = ks * 32;
        // A fragment: 16x32 bf16 (ISA 7.12.2): lane row = lane&15,
        // dword v holds K pair {((v&4)<<2)+((v&3)<<1), +1} + 8*(lane>>4)
        v16bf a;
#pragma unroll
        for (int v = 0; v < 8; ++v) {
            int kp = K0 + ((v & 4) << 2) + ((v & 3) << 1) + (hl << 3);
            unsigned w = *(const unsigned*)&Al[wave * 16 + m][kp];
            put2(a, 2 * v, w);
        }
#pragma unroll
        for (int t = 0; t < NT; ++t) {
            // B fragment: 32x16 bf16: lane col = lane&15,
            // dword v holds K pair {((v&4)<<2)+((v&3)<<1), +1} + 8*(lane>>4)
            v16bf b;
#pragma unroll
            for (int v = 0; v < 8; ++v) {
                int kp = K0 + ((v & 4) << 2) + ((v & 3) << 1) + (hl << 3);
                unsigned w = *(const unsigned*)&Wt[t * 16 + m][kp];
                put2(b, 2 * v, w);
            }
            acc[t] = __builtin_amdgcn_wmma_f32_16x16x32_bf16(
                false, a, false, b, (short)0, acc[t], false, false);
        }
    }

    // Store: C/D layout: vgpr r, lane l -> row = r + 8*(l>>4), col = l&15
    const int rowb = row0 + wave * 16 + 8 * hl;
#pragma unroll
    for (int t = 0; t < NT; ++t) {
        int col = t * 16 + m;
        if (col < ncols) {
#pragma unroll
            for (int r = 0; r < 8; ++r) {
                int row = rowb + r;
                if (row < nrows) {
                    float v = acc[t][r];
                    if (bias) v += bias[col];
                    OUT[(size_t)row * ncols + col] = v;
                }
            }
        }
    }
}

// ---------------------------------------------------------------------------
// Per-node attention logits: es = H . a_src, ed = H . a_dst (one wave / node)
// ---------------------------------------------------------------------------
__global__ __launch_bounds__(256) void logits_kernel(
    const float* __restrict__ H, const float* __restrict__ a_src,
    const float* __restrict__ a_dst, float* __restrict__ es,
    float* __restrict__ ed, int n)
{
    int wave = threadIdx.x >> 5, lane = threadIdx.x & 31;
    int node = blockIdx.x * 8 + wave;
    if (node >= n) return;
    float4 h4 = *(const float4*)(H + (size_t)node * ND + lane * 4);
    float4 s4 = *(const float4*)(a_src + lane * 4);
    float4 d4 = *(const float4*)(a_dst + lane * 4);
    float s = h4.x * s4.x + h4.y * s4.y + h4.z * s4.z + h4.w * s4.w;
    float d = h4.x * d4.x + h4.y * d4.y + h4.z * d4.z + h4.w * d4.w;
#pragma unroll
    for (int o = 16; o > 0; o >>= 1) {
        s += __shfl_down(s, o, 32);
        d += __shfl_down(d, o, 32);
    }
    if (lane == 0) { es[node] = s; ed[node] = d; }
}

// ---------------------------------------------------------------------------
// CSR build over E real edges + N self-loops (appended, matching reference)
// ---------------------------------------------------------------------------
__global__ void count_kernel(const int* __restrict__ ei, int E, int ET,
                             int* __restrict__ deg)
{
    int e = blockIdx.x * blockDim.x + threadIdx.x;
    if (e >= ET) return;
    int d = (e < E) ? ei[E + e] : (e - E);
    atomicAdd(&deg[d], 1);
}

__global__ __launch_bounds__(1024) void scan_kernel(
    const int* __restrict__ deg, int* __restrict__ off,
    int* __restrict__ cursor, int n)
{
    __shared__ int s[1024];
    __shared__ int carry;
    const int tid = threadIdx.x;
    if (tid == 0) carry = 0;
    __syncthreads();
    for (int base = 0; base < n; base += 1024) {
        int i = base + tid;
        int v = (i < n) ? deg[i] : 0;
        s[tid] = v;
        __syncthreads();
        for (int o = 1; o < 1024; o <<= 1) {
            int t = (tid >= o) ? s[tid - o] : 0;
            __syncthreads();
            s[tid] += t;
            __syncthreads();
        }
        int excl = s[tid] - v;
        if (i < n) { off[i] = carry + excl; cursor[i] = carry + excl; }
        __syncthreads();
        if (tid == 1023) carry += s[1023];
        __syncthreads();
    }
    if (tid == 0) off[n] = carry;
}

__global__ void scatter_kernel(const int* __restrict__ ei, int E, int ET,
                               int* __restrict__ cursor, int* __restrict__ csr)
{
    int e = blockIdx.x * blockDim.x + threadIdx.x;
    if (e >= ET) return;
    int d = (e < E) ? ei[E + e] : (e - E);
    int srcn = (e < E) ? ei[e] : (e - E);
    int pos = atomicAdd(&cursor[d], 1);
    csr[pos] = srcn;
}

// Canonicalize each segment (sort by src id) so float accumulation order is
// deterministic across launches despite the atomic scatter.
__global__ void sort_kernel(const int* __restrict__ off, int* __restrict__ csr,
                            int n)
{
    int node = blockIdx.x * blockDim.x + threadIdx.x;
    if (node >= n) return;
    int beg = off[node], end = off[node + 1];
    for (int i = beg + 1; i < end; ++i) {
        int v = csr[i], j = i - 1;
        while (j >= beg && csr[j] > v) { csr[j + 1] = csr[j]; --j; }
        csr[j + 1] = v;
    }
}

// ---------------------------------------------------------------------------
// Segment softmax + weighted aggregation + bias + ReLU (one wave / dst node)
// ---------------------------------------------------------------------------
__global__ __launch_bounds__(256) void aggregate_kernel(
    const float* __restrict__ H, const float* __restrict__ es,
    const float* __restrict__ ed, const int* __restrict__ off,
    const int* __restrict__ csr, const float* __restrict__ bias,
    float* __restrict__ Xout, int n)
{
    int wave = threadIdx.x >> 5, lane = threadIdx.x & 31;
    int node = blockIdx.x * 8 + wave;
    if (node >= n) return;

    int beg = off[node], end = off[node + 1];
    float edd = ed[node];

    // Pass 1: segment max (edges striped over lanes, shfl reduce)
    float mx = -3.4e38f;
    for (int j = beg + lane; j < end; j += 32) {
        float a = es[csr[j]] + edd;
        a = (a > 0.f) ? a : a * NEG_SLOPE;
        mx = fmaxf(mx, a);
    }
#pragma unroll
    for (int o = 16; o > 0; o >>= 1) mx = fmaxf(mx, __shfl_xor(mx, o, 32));

    // Pass 2: exp-sum + weighted gather of h[src] (float4/lane = 512B/row)
    float z = 0.f;
    float4 acc = {0.f, 0.f, 0.f, 0.f};
    for (int j = beg; j < end; ++j) {
        int s = csr[j];
        float a = es[s] + edd;
        a = (a > 0.f) ? a : a * NEG_SLOPE;
        float w = __expf(a - mx);
        z += w;
        float4 h4 = *(const float4*)(H + (size_t)s * ND + lane * 4);
        acc.x += w * h4.x; acc.y += w * h4.y;
        acc.z += w * h4.z; acc.w += w * h4.w;
    }
    float inv = 1.0f / z;
    int c = lane * 4;
    float4 b4 = *(const float4*)(bias + c);
    float4 o;
    o.x = fmaxf(acc.x * inv + b4.x, 0.f);
    o.y = fmaxf(acc.y * inv + b4.y, 0.f);
    o.z = fmaxf(acc.z * inv + b4.z, 0.f);
    o.w = fmaxf(acc.w * inv + b4.w, 0.f);
    *(float4*)(Xout + (size_t)node * ND + c) = o;
}

// ---------------------------------------------------------------------------
extern "C" void kernel_launch(void* const* d_in, const int* in_sizes, int n_in,
                              void* d_out, int out_size, void* d_ws, size_t ws_size,
                              hipStream_t stream)
{
    const float* x  = (const float*)d_in[0];
    const int*   ei = (const int*)d_in[1];
    const int N  = in_sizes[0] / ND;
    const int E  = in_sizes[1] / 2;
    const int ET = E + N;
    const int C  = in_sizes[15];

    char* w = (char*)d_ws;
    size_t o = 0;
    auto alloc = [&](size_t bytes) -> void* {
        void* p = w + o;
        o = (o + bytes + 255) & ~(size_t)255;
        return p;
    };
    float* xbuf = (float*)alloc((size_t)N * ND * 4);
    float* hbuf = (float*)alloc((size_t)N * ND * 4);
    float* es   = (float*)alloc((size_t)N * 4);
    float* ed   = (float*)alloc((size_t)N * 4);
    int* deg    = (int*)alloc((size_t)N * 4);
    int* offp   = (int*)alloc((size_t)(N + 1) * 4);
    int* cursor = (int*)alloc((size_t)N * 4);
    int* csr    = (int*)alloc((size_t)ET * 4);
    (void)ws_size; (void)n_in;

    // --- CSR by destination (done once; reused by all 3 layers) ---
    hipMemsetAsync(deg, 0, (size_t)N * 4, stream);
    count_kernel<<<(ET + 255) / 256, 256, 0, stream>>>(ei, E, ET, deg);
    scan_kernel<<<1, 1024, 0, stream>>>(deg, offp, cursor, N);
    scatter_kernel<<<(ET + 255) / 256, 256, 0, stream>>>(ei, E, ET, cursor, csr);
    sort_kernel<<<(N + 255) / 256, 256, 0, stream>>>(offp, csr, N);

    // --- 3 GAT layers ---
    const float* cur = x;
    for (int l = 0; l < 3; ++l) {
        const float* Wl = (const float*)d_in[2 + 4 * l];
        const float* as = (const float*)d_in[3 + 4 * l];
        const float* ad = (const float*)d_in[4 + 4 * l];
        const float* bl = (const float*)d_in[5 + 4 * l];
        gemm_bf16_wmma<8><<<(N + 63) / 64, 128, 0, stream>>>(
            cur, Wl, nullptr, hbuf, N, ND);
        logits_kernel<<<(N + 7) / 8, 256, 0, stream>>>(hbuf, as, ad, es, ed, N);
        aggregate_kernel<<<(N + 7) / 8, 256, 0, stream>>>(
            hbuf, es, ed, offp, csr, bl, xbuf, N);
        cur = xbuf;
    }

    // --- Head: [N,128] x [128,40] + bias -> d_out ---
    gemm_bf16_wmma<3><<<(N + 63) / 64, 128, 0, stream>>>(
        xbuf, (const float*)d_in[14], (const float*)d_in[15],
        (float*)d_out, N, C);
}
